// GroupQueryAttention_8323646620395
// MI455X (gfx1250) — matrix operand order repaired and checked by reading, not compiled, not run on hardware
//
#include <hip/hip_runtime.h>
#include <math.h>

// ---------------------------------------------------------------------------
// GQA (B=4, S=2048, E=1024, H=16, G=4, HPG=4, hd=64, KV=256) on gfx1250 WMMA.
// All matmuls use v_wmma_f32_16x16x32_bf16 (wave32). f32 inputs converted to
// bf16 once; accumulation and softmax in f32. Softmax reductions are pure
// VALU (DPP row rotations); tile staging uses async global->LDS copies
// (ASYNCcnt) with double-buffered LDS so fills overlap the WMMA bursts.
// ---------------------------------------------------------------------------

typedef __attribute__((ext_vector_type(16))) __bf16 v16bf;
typedef __attribute__((ext_vector_type(8)))  float  v8f;
typedef unsigned int uint32;
typedef int v4i_gcc __attribute__((vector_size(16)));   // matches builtin param

union FragBF { v16bf v; uint32 u[8]; };
union Pack8  { uint4 q; __bf16 e[8]; };

// Softmax runs in the exp2 domain: scale = (1/sqrt(64)) * log2(e), folded into
// the Q projection output.
#define QK_SCALE 0.18033688011112042f

#if defined(__HIP_DEVICE_COMPILE__) && defined(__has_builtin)
#  if __has_builtin(__builtin_amdgcn_exp2f)
#    define FAST_EXP2(x) __builtin_amdgcn_exp2f(x)
#  endif
#  if __has_builtin(__builtin_amdgcn_mov_dpp)
#    define ROW_ROT_MOVDPP 1
#  elif __has_builtin(__builtin_amdgcn_update_dpp)
#    define ROW_ROT_UPDDPP 1
#  endif
#  if __has_builtin(__builtin_amdgcn_global_load_async_to_lds_b128) && \
      __has_builtin(__builtin_amdgcn_s_wait_asynccnt)
#    define HAVE_ASYNC_LDS 1
#  endif
#endif
#ifndef FAST_EXP2
#  define FAST_EXP2(x) exp2f(x)
#endif

// Rotate x by N lanes within each 16-lane row (pure VALU, v_mov_b32_dpp).
// Applying N = 8,4,2,1 with a combining op yields a full 16-lane reduction.
template <int N>
__device__ inline float row_rot16(float x) {
#if defined(ROW_ROT_MOVDPP)
  return __int_as_float(
      __builtin_amdgcn_mov_dpp(__float_as_int(x), 0x120 | N, 0xf, 0xf, true));
#elif defined(ROW_ROT_UPDDPP)
  return __int_as_float(__builtin_amdgcn_update_dpp(
      __float_as_int(x), __float_as_int(x), 0x120 | N, 0xf, 0xf, false));
#else
  return __shfl_xor(x, N, 32);  // xor masks {8,4,2,1} also reduce the row
#endif
}

// 16B global -> LDS copy: async (global_load_async_to_lds_b128, ASYNCcnt) when
// available, else synchronous via VGPRs.
__device__ inline void stage16B(const __bf16* g, __bf16* l) {
#ifdef HAVE_ASYNC_LDS
  __builtin_amdgcn_global_load_async_to_lds_b128(
      (__attribute__((address_space(1))) v4i_gcc*)g,
      (__attribute__((address_space(3))) v4i_gcc*)l, 0, 0);
#else
  *(uint4*)l = *(const uint4*)g;
#endif
}

__device__ inline void stage_wait() {
#ifdef HAVE_ASYNC_LDS
  __builtin_amdgcn_s_wait_asynccnt(0);  // barrier does not cover ASYNCcnt
#endif
}

__device__ inline v8f splat8(float x) {
  v8f r;
#pragma unroll
  for (int i = 0; i < 8; ++i) r[i] = x;
  return r;
}

// ---------------------------------------------------------------------------
// Elementwise f32 -> bf16 convert (for x).
// ---------------------------------------------------------------------------
__global__ void cvt_f32_to_bf16(const float* __restrict__ src,
                                __bf16* __restrict__ dst, int n) {
  for (int i = blockIdx.x * blockDim.x + threadIdx.x; i < n;
       i += gridDim.x * blockDim.x)
    dst[i] = (__bf16)src[i];
}

// W[k][n] (KxN row-major) -> WT[n][k] bf16 (N-major) so GEMM B-fragments read
// contiguous K pairs.
__global__ void cvt_transpose_bf16(const float* __restrict__ W,
                                   __bf16* __restrict__ WT, int K, int N) {
  int total = K * N;
  for (int i = blockIdx.x * blockDim.x + threadIdx.x; i < total;
       i += gridDim.x * blockDim.x) {
    int n = i / K, k = i - n * K;
    WT[i] = (__bf16)W[(size_t)k * N + n];
  }
}

// ---------------------------------------------------------------------------
// bf16 GEMM: C[M,N] = (A[M,K] * BT[N,K]^T + bias[N]) * out_scale.
// Block tile 128x64, 8 waves (4x2), each wave 32x32 (2x2 WMMA tiles).
// Double-buffered async LDS staging, K-step 32.
// ---------------------------------------------------------------------------
#define LSA_STRIDE 40   // 128 rows * 40 bf16 (80B rows, 16B aligned)
#define LSB_STRIDE 40

template <bool F32OUT>
__global__ __launch_bounds__(256)
void gemm_bf16_wmma(const __bf16* __restrict__ A, const __bf16* __restrict__ BT,
                    const float* __restrict__ bias,
                    __bf16* __restrict__ Cb, float* __restrict__ Cf,
                    int M, int N, int Kd, float out_scale) {
  __shared__ __bf16 lsa[2][128 * LSA_STRIDE];
  __shared__ __bf16 lsb[2][64 * LSB_STRIDE];
  (void)M;

  const int t    = threadIdx.x;
  const int wave = t >> 5;
  const int lane = t & 31;
  const int lo   = lane & 15;
  const int h    = lane >> 4;
  const int wm   = wave >> 1;          // 0..3 : wave row
  const int wn   = wave & 1;           // 0..1 : wave col
  const int Mb   = blockIdx.x * 128;
  const int Nb   = blockIdx.y * 64;

  v8f acc[2][2];
#pragma unroll
  for (int nt = 0; nt < 2; ++nt) {
    float bv = bias[Nb + wn * 32 + nt * 16 + lo];
#pragma unroll
    for (int mt = 0; mt < 2; ++mt) acc[mt][nt] = splat8(bv);
  }

  const int arow = t >> 1, ahalf = t & 1;   // A: 128 rows x 32 cols, 16 bf16/thr
  const int brow = t >> 2, bpart = t & 3;   // B: 64 rows x 32 cols,  8 bf16/thr
  const __bf16* gA = A  + (size_t)(Mb + arow) * Kd + ahalf * 16;
  const __bf16* gB = BT + (size_t)(Nb + brow) * Kd + bpart * 8;

  auto stageAB = [&](int k0, int bufi) {
    __bf16* la = &lsa[bufi][arow * LSA_STRIDE + ahalf * 16];
    __bf16* lb = &lsb[bufi][brow * LSB_STRIDE + bpart * 8];
    stage16B(gA + k0,     la);
    stage16B(gA + k0 + 8, la + 8);
    stage16B(gB + k0,     lb);
  };

  stageAB(0, 0);          // prologue fill
  stage_wait();
  __syncthreads();

  for (int k0 = 0; k0 < Kd; k0 += 32) {
    const int cur = (k0 >> 5) & 1;
    if (k0 + 32 < Kd) {
      stageAB(k0 + 32, cur ^ 1);                // async fill of next buffer
      if (k0 + 64 < Kd) {                       // warm L2 one more tile ahead
        __builtin_prefetch(gA + k0 + 64, 0, 0);
        __builtin_prefetch(gB + k0 + 64, 0, 0);
      }
    }

    FragBF fa[2], fb[2];
#pragma unroll
    for (int mt = 0; mt < 2; ++mt) {
      const __bf16* base = &lsa[cur][(wm * 32 + mt * 16 + lo) * LSA_STRIDE];
#pragma unroll
      for (int j = 0; j < 8; ++j) {
        // A fragment, this code's k order: lane half h holds K = h*16 + 0..15
        int koff = h * 16 + 2 * j;
        fa[mt].u[j] = *(const uint32*)(base + koff);
      }
    }
#pragma unroll
    for (int nt = 0; nt < 2; ++nt) {
      // B fragment, same k order as A: lane half h holds K = h*16 + 0..15
      const __bf16* base =
          &lsb[cur][(wn * 32 + nt * 16 + lo) * LSB_STRIDE + h * 16];
#pragma unroll
      for (int j = 0; j < 8; ++j) fb[nt].u[j] = *(const uint32*)(base + 2 * j);
    }

#pragma unroll
    for (int mt = 0; mt < 2; ++mt)
#pragma unroll
      for (int nt = 0; nt < 2; ++nt)
        acc[mt][nt] = __builtin_amdgcn_wmma_f32_16x16x32_bf16(
            false, fa[mt].v, false, fb[nt].v, (short)0, acc[mt][nt],
            false, false);

    stage_wait();     // next buffer's async writes landed
    __syncthreads();  // everyone done reading cur; safe to overwrite next iter
  }

  // C/D layout: component r -> M = h*8 + r, N = lo
#pragma unroll
  for (int mt = 0; mt < 2; ++mt) {
#pragma unroll
    for (int nt = 0; nt < 2; ++nt) {
      const int col = Nb + wn * 32 + nt * 16 + lo;
#pragma unroll
      for (int r = 0; r < 8; ++r) {
        const int row = Mb + wm * 32 + mt * 16 + h * 8 + r;
        const float v = acc[mt][nt][r] * out_scale;
        if (F32OUT) Cf[(size_t)row * N + col] = v;
        else        Cb[(size_t)row * N + col] = (__bf16)v;
      }
    }
  }
}

// ---------------------------------------------------------------------------
// Flash attention per (b, g). Query rows r in [0, HPG*S) with p = r>>11,
// s = r&2047, all sharing K/V "head" g. Block = 128 rows (8 waves x 16),
// streaming 64-key tiles with online softmax (exp2 domain; Q pre-scaled),
// double-buffered K/V staging. Output written in the reference concat order:
// col = p*256 + g*64 + d.
// ---------------------------------------------------------------------------
#define SK 72   // K tile stride  (144B rows, 16B aligned)
#define SV 72   // V^T tile stride
#define SP 72   // per-wave P tile stride

__global__ __launch_bounds__(256)
void gqa_flash_wmma(const __bf16* __restrict__ Qb, const __bf16* __restrict__ Kb,
                    const __bf16* __restrict__ Vb, __bf16* __restrict__ Ob) {
  __shared__ __bf16 lsk[2][64 * SK];       // K tile  [key][d]
  __shared__ __bf16 lsv[2][64 * SV];       // V tile transposed [d][key]
  __shared__ __bf16 lsp[8 * 16 * SP];      // per-wave P [qrow][key]

  const int t    = threadIdx.x;
  const int wave = t >> 5;
  const int lane = t & 31;
  const int lo   = lane & 15;
  const int h    = lane >> 4;
  const int b    = blockIdx.z;
  const int g    = blockIdx.y;
  const int rbase = blockIdx.x * 128 + wave * 16;

  // Q fragments (constant over key loop). A-layout: lane lo = query row.
  FragBF qa[2];
  {
    const int rq = rbase + lo;
    const int p = rq >> 11, s = rq & 2047;
    const __bf16* qp = Qb + ((size_t)(b * 2048 + s) * 16 + (g * 4 + p)) * 64;
#pragma unroll
    for (int f = 0; f < 2; ++f)
#pragma unroll
      for (int j = 0; j < 8; ++j) {
        int d = f * 32 + h * 16 + 2 * j;
        qa[f].u[j] = *(const uint32*)(qp + d);
      }
  }

  float m_run[8], l_run[8];
  v8f O[4];
#pragma unroll
  for (int r = 0; r < 8; ++r) { m_run[r] = -1e30f; l_run[r] = 0.0f; }
#pragma unroll
  for (int dt = 0; dt < 4; ++dt) O[dt] = splat8(0.0f);

  const int skey = t >> 2, spart = t & 3;   // staging: 64 keys x 4 thr, 16 bf16

  auto stageKV = [&](int kt, int bufi) {
    const int key = kt + skey;
    const size_t base = ((size_t)(b * 2048 + key) * 4 + g) * 64 + spart * 16;
    // K tile: async straight into LDS (natural [key][d] layout)
    stage16B(Kb + base,     &lsk[bufi][skey * SK + spart * 16]);
    stage16B(Kb + base + 8, &lsk[bufi][skey * SK + spart * 16 + 8]);
    // V staged transposed: lsv[d][key] (manual; async cannot transpose)
    Pack8 v0, v1;
    const uint4* gv = (const uint4*)(Vb + base);
    v0.q = gv[0]; v1.q = gv[1];
#pragma unroll
    for (int e = 0; e < 8; ++e) {
      lsv[bufi][(spart * 16 + e    ) * SV + skey] = v0.e[e];
      lsv[bufi][(spart * 16 + e + 8) * SV + skey] = v1.e[e];
    }
    if (kt + 64 < 2048) {                     // warm L2 one tile further ahead
      __builtin_prefetch(Kb + base + 4 * 64 * 64, 0, 0);
      __builtin_prefetch(Vb + base + 4 * 64 * 64, 0, 0);
    }
  };

  stageKV(0, 0);          // prologue fill
  stage_wait();
  __syncthreads();

  for (int kt = 0; kt < 2048; kt += 64) {
    const int cur = (kt >> 6) & 1;
    if (kt + 64 < 2048) stageKV(kt + 64, cur ^ 1);  // fill next buffer (async)

    // scores: S[16 q][64 keys] = Q(16x64) * K^T(64x64)  (already exp2-scaled)
    v8f Sc[4];
#pragma unroll
    for (int nt = 0; nt < 4; ++nt) {
      const __bf16* kb = &lsk[cur][(nt * 16 + lo) * SK + h * 16];
      FragBF f0, f1;
#pragma unroll
      for (int j = 0; j < 8; ++j) {
        f0.u[j] = *(const uint32*)(kb + 2 * j);        // d = h*16 + i
        f1.u[j] = *(const uint32*)(kb + 32 + 2 * j);   // d = 32 + h*16 + i
      }
      v8f z = splat8(0.0f);
      z = __builtin_amdgcn_wmma_f32_16x16x32_bf16(false, qa[0].v, false, f0.v,
                                                  (short)0, z, false, false);
      z = __builtin_amdgcn_wmma_f32_16x16x32_bf16(false, qa[1].v, false, f1.v,
                                                  (short)0, z, false, false);
      Sc[nt] = z;
    }

    // online softmax; row m = h*8 + r lives across the 16 lanes sharing h.
    // All reductions are VALU DPP rotations (no LDS, no DS waits).
#pragma unroll
    for (int r = 0; r < 8; ++r) {
      float mx = fmaxf(fmaxf(Sc[0][r], Sc[1][r]), fmaxf(Sc[2][r], Sc[3][r]));
      mx = fmaxf(mx, row_rot16<8>(mx));
      mx = fmaxf(mx, row_rot16<4>(mx));
      mx = fmaxf(mx, row_rot16<2>(mx));
      mx = fmaxf(mx, row_rot16<1>(mx));
      const float mn = fmaxf(m_run[r], mx);
      const float alpha = FAST_EXP2(m_run[r] - mn);
      m_run[r] = mn;
      float rs = 0.0f;
#pragma unroll
      for (int nt = 0; nt < 4; ++nt) {
        float pv = FAST_EXP2(Sc[nt][r] - mn);
        Sc[nt][r] = pv;
        rs += pv;
      }
      rs += row_rot16<8>(rs);
      rs += row_rot16<4>(rs);
      rs += row_rot16<2>(rs);
      rs += row_rot16<1>(rs);
      l_run[r] = l_run[r] * alpha + rs;
#pragma unroll
      for (int dt = 0; dt < 4; ++dt) O[dt][r] *= alpha;
    }

    // C-layout -> A-layout for P via per-wave LDS round trip
    __bf16* pw = &lsp[wave * 16 * SP];
#pragma unroll
    for (int nt = 0; nt < 4; ++nt)
#pragma unroll
      for (int r = 0; r < 8; ++r)
        pw[(h * 8 + r) * SP + nt * 16 + lo] = (__bf16)Sc[nt][r];

    // O += P(16x64) * V(64x64): contraction over keys, two k-steps of 32
#pragma unroll
    for (int ks = 0; ks < 2; ++ks) {
      FragBF pa;
#pragma unroll
      for (int j = 0; j < 8; ++j) {
        int key = ks * 32 + h * 16 + 2 * j;
        pa.u[j] = *(const uint32*)(pw + lo * SP + key);
      }
#pragma unroll
      for (int dt = 0; dt < 4; ++dt) {
        const __bf16* vb = &lsv[cur][(dt * 16 + lo) * SV + ks * 32 + h * 16];
        FragBF fv;
#pragma unroll
        for (int j = 0; j < 8; ++j) fv.u[j] = *(const uint32*)(vb + 2 * j);
        O[dt] = __builtin_amdgcn_wmma_f32_16x16x32_bf16(
            false, pa.v, false, fv.v, (short)0, O[dt], false, false);
      }
    }

    stage_wait();     // next buffer's async K writes landed
    __syncthreads();  // everyone done reading cur; V ds-stores covered too
  }

  // normalize and store in concat order col = p*256 + g*64 + d
#pragma unroll
  for (int r = 0; r < 8; ++r) {
    const int rq = rbase + h * 8 + r;
    const int p = rq >> 11, s = rq & 2047;
    const float inv = 1.0f / l_run[r];
    __bf16* op = Ob + (size_t)(b * 2048 + s) * 1024 + p * 256 + g * 64;
#pragma unroll
    for (int dt = 0; dt < 4; ++dt)
      op[dt * 16 + lo] = (__bf16)(O[dt][r] * inv);
  }
}

// ---------------------------------------------------------------------------
// Host-side orchestration.
// ---------------------------------------------------------------------------
extern "C" void kernel_launch(void* const* d_in, const int* in_sizes, int n_in,
                              void* d_out, int out_size, void* d_ws,
                              size_t ws_size, hipStream_t stream) {
  (void)in_sizes; (void)n_in; (void)out_size; (void)ws_size;
  constexpr int Bb = 4, Ss = 2048, Ee = 1024, Gg = 4, KV = 256;
  constexpr size_t ME = (size_t)Bb * Ss;  // 8192 token rows

  const float* x  = (const float*)d_in[0];
  const float* Wq = (const float*)d_in[1];
  const float* bq = (const float*)d_in[2];
  const float* Wk = (const float*)d_in[3];
  const float* bk = (const float*)d_in[4];
  const float* Wv = (const float*)d_in[5];
  const float* bv = (const float*)d_in[6];
  const float* Wo = (const float*)d_in[7];
  const float* bo = (const float*)d_in[8];
  float* out = (float*)d_out;

  // Workspace layout (bf16 elements), ~45 MB total.
  __bf16* ws = (__bf16*)d_ws;
  size_t o = 0;
  __bf16* xb  = ws + o; o += ME * Ee;            // x bf16; reused for attn out
  __bf16* WqT = ws + o; o += (size_t)Ee * Ee;    // Wq^T
  __bf16* WkT = ws + o; o += (size_t)Ee * KV;
  __bf16* WvT = ws + o; o += (size_t)Ee * KV;
  __bf16* WoT = ws + o; o += (size_t)Ee * Ee;
  __bf16* Qb  = ws + o; o += ME * Ee;
  __bf16* Kb  = ws + o; o += ME * KV;
  __bf16* Vb  = ws + o; o += ME * KV;
  __bf16* AOb = xb;  // x is dead after the Q/K/V GEMMs

  cvt_f32_to_bf16<<<2048, 256, 0, stream>>>(x, xb, (int)(ME * Ee));
  cvt_transpose_bf16<<<1024, 256, 0, stream>>>(Wq, WqT, Ee, Ee);
  cvt_transpose_bf16<<<256, 256, 0, stream>>>(Wk, WkT, Ee, KV);
  cvt_transpose_bf16<<<256, 256, 0, stream>>>(Wv, WvT, Ee, KV);
  cvt_transpose_bf16<<<1024, 256, 0, stream>>>(Wo, WoT, Ee, Ee);

  // Q projection folds in (1/sqrt(hd)) * log2(e) for exp2-domain softmax.
  gemm_bf16_wmma<false><<<dim3(ME / 128, Ee / 64), 256, 0, stream>>>(
      xb, WqT, bq, Qb, nullptr, (int)ME, Ee, Ee, QK_SCALE);
  gemm_bf16_wmma<false><<<dim3(ME / 128, KV / 64), 256, 0, stream>>>(
      xb, WkT, bk, Kb, nullptr, (int)ME, KV, Ee, 1.0f);
  gemm_bf16_wmma<false><<<dim3(ME / 128, KV / 64), 256, 0, stream>>>(
      xb, WvT, bv, Vb, nullptr, (int)ME, KV, Ee, 1.0f);

  gqa_flash_wmma<<<dim3(64, Gg, Bb), 256, 0, stream>>>(Qb, Kb, Vb, AOb);

  gemm_bf16_wmma<true><<<dim3(ME / 128, Ee / 64), 256, 0, stream>>>(
      AOb, WoT, bo, nullptr, out, (int)ME, Ee, Ee, 1.0f);
}
